// SelfAttention_32040456028329
// MI455X (gfx1250) — matrix operand order repaired and checked by reading, not compiled, not run on hardware
//
#include <hip/hip_runtime.h>
#include <hip/hip_bf16.h>
#include <stdint.h>

typedef unsigned short u16;
typedef __attribute__((ext_vector_type(16))) __bf16 bf16x16;
typedef __attribute__((ext_vector_type(8)))  __bf16 bf16x8;
typedef __attribute__((ext_vector_type(8)))  float  f32x8;

#define BATCH 2
#define SEQ   2048
#define CH    1024
#define NH    16
#define HD    64
#define MROWS (BATCH*SEQ)   // 4096
#define BM    128
#define BN    128
#define BK    32

__device__ __forceinline__ u16 f2bf(float f) {
  unsigned u = __builtin_bit_cast(unsigned, f);
  u += 0x7FFFu + ((u >> 16) & 1u);   // round-to-nearest-even
  return (u16)(u >> 16);
}

__device__ __forceinline__ bf16x16 join8(bf16x8 lo, bf16x8 hi) {
  return __builtin_shufflevector(lo, hi, 0,1,2,3,4,5,6,7,8,9,10,11,12,13,14,15);
}

// async DMA: 16 bytes global -> LDS, tracked by ASYNCcnt (no VGPR round-trip)
__device__ __forceinline__ void async_cp16(const u16* g, u16* l) {
  unsigned lds_off = (unsigned)(uintptr_t)l;       // low 32 bits = LDS offset
  unsigned long long ga = (unsigned long long)(uintptr_t)g;
  asm volatile("global_load_async_to_lds_b128 %0, %1, off"
               :: "v"(lds_off), "v"(ga) : "memory");
}
__device__ __forceinline__ void wait_async_le(int) {}  // (doc helper, see asm below)

// ---------------------------------------------------------------- convert
__global__ void cvt_f32_to_bf16(const float* __restrict__ in, u16* __restrict__ out) {
  int i = (blockIdx.x * blockDim.x + threadIdx.x) * 4;
  float4 v = *(const float4*)(in + i);
  unsigned lo = (unsigned)f2bf(v.x) | ((unsigned)f2bf(v.y) << 16);
  unsigned hi = (unsigned)f2bf(v.z) | ((unsigned)f2bf(v.w) << 16);
  *(uint2*)(out + i) = make_uint2(lo, hi);
}

// ---------------------------------------------------------------- GEMM: A[M,K] @ W[N,K]^T (+bias)
// 128x128 block tile, 8 waves, 32x64 wave tile (8 WMMA / K-step),
// double-buffered LDS fed by global_load_async_to_lds_b128.
// mode 0: write bf16 to [B,H,T,D] qkv layout;  mode 1: write fp32 [M,N]
__global__ __launch_bounds__(256)
void gemm_xwT(const u16* __restrict__ A, const u16* __restrict__ W,
              const float* __restrict__ bias,
              u16* __restrict__ out_qkv, float* __restrict__ out_f32,
              int mode) {
  __shared__ alignas(16) u16 Alds[2][BM*BK];   // 2 x 8KB
  __shared__ alignas(16) u16 Blds[2][BN*BK];   // 2 x 8KB
  const int tid  = threadIdx.x;
  const int lane = tid & 31;
  const int wv   = tid >> 5;
  const int wm   = wv >> 1;          // 0..3 : 32-row strip
  const int wn   = wv & 1;           // 0..1 : 64-col strip
  const int row0 = blockIdx.y * BM;
  const int n0   = blockIdx.x * BN;
  const int hi16 = lane >> 4;
  const int l16  = lane & 15;

  f32x8 acc[2][4] = {};

  // per-thread: 2 A-chunks + 2 B-chunks = 4 async ops per stage
  auto stage = [&](int kk, int s) {
    #pragma unroll
    for (int c = 0; c < 2; ++c) {
      int j = tid + c*256;           // 0..511
      int r = j >> 2, q = j & 3;
      async_cp16(&A[(size_t)(row0 + r)*CH + kk + q*8], &Alds[s][r*BK + q*8]);
      async_cp16(&W[(size_t)(n0  + r)*CH + kk + q*8], &Blds[s][r*BK + q*8]);
    }
  };

  stage(0, 0);
  int s = 0;
  for (int kk = 0; kk < CH; kk += BK) {
    if (kk + BK < CH) {
      stage(kk + BK, s ^ 1);                              // prefetch next tile
      asm volatile("s_wait_asynccnt 0x4" ::: "memory");   // current tile landed
    } else {
      asm volatile("s_wait_asynccnt 0x0" ::: "memory");
    }
    __syncthreads();                                      // publish across waves

    bf16x16 afrag[2], bfrag[4];
    #pragma unroll
    for (int i = 0; i < 2; ++i) {
      int row = wm*32 + i*16 + l16;
      int c0  = hi16 ? 8 : 0;
      bf16x8 lo = *(const bf16x8*)&Alds[s][row*BK + c0];
      bf16x8 hi = *(const bf16x8*)&Alds[s][row*BK + c0 + 16];
      afrag[i] = join8(lo, hi);
    }
    #pragma unroll
    for (int j = 0; j < 4; ++j) {
      int n  = wn*64 + j*16 + l16;
      int k0 = hi16 ? 8 : 0;
      bf16x8 lo = *(const bf16x8*)&Blds[s][n*BK + k0];
      bf16x8 hi = *(const bf16x8*)&Blds[s][n*BK + k0 + 16];
      bfrag[j] = join8(lo, hi);
    }
    #pragma unroll
    for (int i = 0; i < 2; ++i)
      #pragma unroll
      for (int j = 0; j < 4; ++j)
        acc[i][j] = __builtin_amdgcn_wmma_f32_16x16x32_bf16(
            false, afrag[i], false, bfrag[j], (short)0, acc[i][j], false, false);
    __syncthreads();                 // all waves done before buffer reuse
    s ^= 1;
  }

  #pragma unroll
  for (int i = 0; i < 2; ++i) {
    #pragma unroll
    for (int j = 0; j < 4; ++j) {
      int coln = n0 + wn*64 + j*16 + l16;
      float bvv = bias ? bias[coln] : 0.0f;
      #pragma unroll
      for (int r = 0; r < 8; ++r) {
        int row = row0 + wm*32 + i*16 + r + 8*hi16;
        float v = acc[i][j][r] + bvv;
        if (mode == 0) {
          int b = row >> 11, t = row & (SEQ - 1);
          int h = coln >> 6, d = coln & (HD - 1);
          out_qkv[(((size_t)(b*NH + h))*SEQ + t)*HD + d] = f2bf(v);
        } else {
          out_f32[(size_t)row*CH + coln] = v;
        }
      }
    }
  }
}

// ---------------------------------------------------------------- flash attention
// grid (SEQ/64, B*H), 128 threads = 4 waves; wave owns a 16-row query strip
__global__ __launch_bounds__(128)
void attn_fwd(const u16* __restrict__ Q, const u16* __restrict__ K,
              const u16* __restrict__ V, u16* __restrict__ Y) {
  __shared__ alignas(16) u16 kt[32*64];     // [key][d]
  __shared__ alignas(16) u16 vt[64*32];     // [d][key] (transposed)
  __shared__ alignas(16) u16 pl[4*16*32];   // per-wave P scratch
  const int tid   = threadIdx.x;
  const int lane  = tid & 31;
  const int wv    = tid >> 5;
  const int hi16  = lane >> 4;
  const int l16   = lane & 15;
  const int bh    = blockIdx.y;             // b*16 + h
  const int qbase = blockIdx.x * 64;
  const int qrow0 = qbase + wv*16;
  const size_t hb = (size_t)bh * SEQ * HD;

  const float SL = 0.125f * 1.44269504088896340736f;  // D^-0.5 * log2(e)

  bf16x16 qa[2];
  {
    int mr = qrow0 + l16;
    int c0 = hi16 ? 8 : 0;
    #pragma unroll
    for (int t = 0; t < 2; ++t) {
      int d0 = t*32;
      bf16x8 lo = *(const bf16x8*)&Q[hb + (size_t)mr*HD + d0 + c0];
      bf16x8 hi = *(const bf16x8*)&Q[hb + (size_t)mr*HD + d0 + c0 + 16];
      qa[t] = join8(lo, hi);
    }
  }

  f32x8 o[4] = {};
  float rm[8], rl[8];
  #pragma unroll
  for (int r = 0; r < 8; ++r) { rm[r] = -1.0e30f; rl[r] = 0.0f; }

  const int kend = qbase + 64;
  for (int kb0 = 0; kb0 < kend; kb0 += 32) {
    // K tile: async DMA straight into LDS (contiguous 4KB, 2 chunks/thread)
    #pragma unroll
    for (int j = tid; j < 256; j += 128)
      async_cp16(&K[hb + (size_t)kb0*HD + j*8], &kt[j*8]);
    // V tile: manual transposed staging [d][key]
    #pragma unroll 4
    for (int j = tid; j < 2048; j += 128) {
      int key = j >> 6, d = j & 63;
      vt[d*32 + key] = V[hb + (size_t)(kb0 + key)*HD + d];
    }
    asm volatile("s_wait_asynccnt 0x0" ::: "memory");
    __syncthreads();

    // S = Q K^T : two 16-col subtiles, contraction over d
    f32x8 s[2];
    #pragma unroll
    for (int sub = 0; sub < 2; ++sub) {
      f32x8 c = {};
      #pragma unroll
      for (int t = 0; t < 2; ++t) {
        int d0 = t*32;
        int n  = sub*16 + l16;
        int k0 = hi16 ? 8 : 0;
        bf16x8 lo = *(const bf16x8*)&kt[n*64 + d0 + k0];
        bf16x8 hi = *(const bf16x8*)&kt[n*64 + d0 + k0 + 16];
        bf16x16 kf = join8(lo, hi);
        c = __builtin_amdgcn_wmma_f32_16x16x32_bf16(
                false, qa[t], false, kf, (short)0, c, false, false);
      }
      s[sub] = c;
    }

    // online softmax (exp2 domain); causal mask; half-wave row reductions
    float p0[8], p1[8], alpha[8];
    #pragma unroll
    for (int r = 0; r < 8; ++r) {
      int trow = qrow0 + r + 8*hi16;
      float v0 = (kb0 + l16      <= trow) ? s[0][r]*SL : -1.0e30f;
      float v1 = (kb0 + 16 + l16 <= trow) ? s[1][r]*SL : -1.0e30f;
      float mx = fmaxf(v0, v1);
      #pragma unroll
      for (int off = 1; off < 16; off <<= 1)
        mx = fmaxf(mx, __shfl_xor(mx, off, 16));
      float mnew = fmaxf(rm[r], mx);
      float a  = __builtin_amdgcn_exp2f(rm[r] - mnew);
      float e0 = __builtin_amdgcn_exp2f(v0 - mnew);
      float e1 = __builtin_amdgcn_exp2f(v1 - mnew);
      float sum = e0 + e1;
      #pragma unroll
      for (int off = 1; off < 16; off <<= 1)
        sum += __shfl_xor(sum, off, 16);
      rl[r] = rl[r]*a + sum;
      rm[r] = mnew;
      alpha[r] = a;
      p0[r] = e0; p1[r] = e1;
    }
    #pragma unroll
    for (int j = 0; j < 4; ++j)
      #pragma unroll
      for (int r = 0; r < 8; ++r)
        o[j][r] *= alpha[r];

    // P: accumulator layout -> A layout via per-wave LDS region
    #pragma unroll
    for (int r = 0; r < 8; ++r) {
      int prow = r + 8*hi16;
      pl[wv*512 + prow*32 + l16]      = f2bf(p0[r]);
      pl[wv*512 + prow*32 + 16 + l16] = f2bf(p1[r]);
    }
    asm volatile("s_wait_dscnt 0" ::: "memory");

    bf16x16 pf;
    {
      int c0 = hi16 ? 8 : 0;
      bf16x8 lo = *(const bf16x8*)&pl[wv*512 + l16*32 + c0];
      bf16x8 hi = *(const bf16x8*)&pl[wv*512 + l16*32 + c0 + 16];
      pf = join8(lo, hi);
    }

    // O += P V  (contraction over 32 keys; V^T in LDS so lanes read contiguous)
    #pragma unroll
    for (int j = 0; j < 4; ++j) {
      int d  = j*16 + l16;
      int k0 = hi16 ? 8 : 0;
      bf16x8 lo = *(const bf16x8*)&vt[d*32 + k0];
      bf16x8 hi = *(const bf16x8*)&vt[d*32 + k0 + 16];
      bf16x16 vf = join8(lo, hi);
      o[j] = __builtin_amdgcn_wmma_f32_16x16x32_bf16(
                 false, pf, false, vf, (short)0, o[j], false, false);
    }
    __syncthreads();
  }

  // epilogue: y in [B,T,C] bf16 for the output projection
  const int b = bh >> 4, h = bh & 15;
  #pragma unroll
  for (int r = 0; r < 8; ++r) {
    float inv = 1.0f / rl[r];
    int t = qrow0 + r + 8*hi16;
    size_t base = ((size_t)b*SEQ + t)*CH + h*HD;
    #pragma unroll
    for (int j = 0; j < 4; ++j)
      Y[base + j*16 + l16] = f2bf(o[j][r] * inv);
  }
}

// ---------------------------------------------------------------- launch
extern "C" void kernel_launch(void* const* d_in, const int* in_sizes, int n_in,
                              void* d_out, int out_size, void* d_ws, size_t ws_size,
                              hipStream_t stream) {
  (void)in_sizes; (void)n_in; (void)out_size; (void)ws_size;
  const float* x  = (const float*)d_in[0];
  const float* Wk = (const float*)d_in[1];
  const float* Wq = (const float*)d_in[2];
  const float* bq = (const float*)d_in[3];
  const float* Wv = (const float*)d_in[4];
  const float* bv = (const float*)d_in[5];
  const float* Wo = (const float*)d_in[6];
  const float* bo = (const float*)d_in[7];
  float* out = (float*)d_out;

  u16* p = (u16*)d_ws;
  const size_t NX = (size_t)MROWS*CH;   // 4M elems
  const size_t NW = (size_t)CH*CH;      // 1M elems
  u16* xb  = p;  p += NX;
  u16* wqb = p;  p += NW;
  u16* wkb = p;  p += NW;
  u16* wvb = p;  p += NW;
  u16* wob = p;  p += NW;
  u16* qb  = p;  p += NX;
  u16* kb  = p;  p += NX;
  u16* vb  = p;  p += NX;
  u16* yb  = p;  p += NX;

  cvt_f32_to_bf16<<<NX/1024, 256, 0, stream>>>(x,  xb);
  cvt_f32_to_bf16<<<NW/1024, 256, 0, stream>>>(Wq, wqb);
  cvt_f32_to_bf16<<<NW/1024, 256, 0, stream>>>(Wk, wkb);
  cvt_f32_to_bf16<<<NW/1024, 256, 0, stream>>>(Wv, wvb);
  cvt_f32_to_bf16<<<NW/1024, 256, 0, stream>>>(Wo, wob);

  dim3 ggrid(CH/BN, MROWS/BM);          // (8, 32)
  gemm_xwT<<<ggrid, 256, 0, stream>>>(xb, wqb, bq,      qb, nullptr, 0);
  gemm_xwT<<<ggrid, 256, 0, stream>>>(xb, wkb, nullptr, kb, nullptr, 0);
  gemm_xwT<<<ggrid, 256, 0, stream>>>(xb, wvb, bv,      vb, nullptr, 0);

  attn_fwd<<<dim3(SEQ/64, BATCH*NH), 128, 0, stream>>>(qb, kb, vb, yb);

  gemm_xwT<<<ggrid, 256, 0, stream>>>(yb, wob, bo, nullptr, out, 1);
}
